// CLLADetect_10926396801207
// MI455X (gfx1250) — matrix operand order repaired and checked by reading, not compiled, not run on hardware
//
#include <hip/hip_runtime.h>
#include <hip/hip_bf16.h>
#include <math.h>

// ---------------------------------------------------------------------------
// CLLADetect forward for MI455X (gfx1250, wave32, WMMA + async LDS copies).
//
// Heavy math: v_wmma_f32_16x16x32_bf16 (f32 accumulate). f1 is never
// materialized (wk/wv folded with w_conv1). Intermediates f2/q/k/v/f are
// bf16 NHWC (halves HBM traffic vs f32; operands are bf16 for WMMA anyway).
// bf16-input GEMMs stage their B tile with global_load_async_to_lds_b128
// (ASYNCcnt) -- pure byte copy, no VGPR round trip. Double-buffered LDS,
// one barrier per K chunk. Head outputs permuted to [b,3,ny,nx,85] in the
// GEMM epilogue (f32, straight into d_out).
// ---------------------------------------------------------------------------

typedef __attribute__((ext_vector_type(16))) __bf16 v16bf;
typedef __attribute__((ext_vector_type(8)))  float  v8f;

__device__ __forceinline__ unsigned short f32_to_bf16_rne(float f) {
    unsigned int u = __builtin_bit_cast(unsigned int, f);
    u += 0x7FFFu + ((u >> 16) & 1u);          // round-to-nearest-even
    return (unsigned short)(u >> 16);
}
__device__ __forceinline__ uint2 pack4_bf16(float4 v) {
    uint2 r;
    r.x = (unsigned)f32_to_bf16_rne(v.x) | ((unsigned)f32_to_bf16_rne(v.y) << 16);
    r.y = (unsigned)f32_to_bf16_rne(v.z) | ((unsigned)f32_to_bf16_rne(v.w) << 16);
    return r;
}
// unpack packed 2xbf16 (low, high) to floats
__device__ __forceinline__ void unpack2_bf16(unsigned u, float& lo, float& hi) {
    lo = __builtin_bit_cast(float, u << 16);
    hi = __builtin_bit_cast(float, u & 0xFFFF0000u);
}

// ---------------------------------------------------------------------------
// Weight folding: wOut = wA @ wB (both [128,128] [out,in]), bOut = wA@bB + bA
// ---------------------------------------------------------------------------
__global__ __launch_bounds__(128) void fuse_weights(
    const float* __restrict__ wA, const float* __restrict__ wB,
    const float* __restrict__ bA, const float* __restrict__ bB,
    float* __restrict__ wOut, float* __restrict__ bOut) {
    int o = blockIdx.x, c = threadIdx.x;
    float s = 0.f;
    for (int m = 0; m < 128; ++m) s += wA[o * 128 + m] * wB[m * 128 + c];
    wOut[o * 128 + c] = s;
    if (c == 0) {
        float sb = bA[o];
        for (int m = 0; m < 128; ++m) sb += wA[o * 128 + m] * bB[m];
        bOut[o] = sb;
    }
}

// ---------------------------------------------------------------------------
// Tiled bf16 WMMA GEMM:  Y[b,o,n] = sum_c W[o,c] * X[...] + bias[o]
//   INMODE 0 : X f32  [b, C, N]  (raw conv inputs, convert while staging)
//   INMODE 1 : X bf16 [b, N, C]  (intermediates; async b128 copy to LDS)
//   OUTMODE 0: Y bf16 [b, N, O]  (packed b128 stores)
//   OUTMODE 1: Y f32 head: Y[b, m/85, n, m%85] over [b,3,N,85]
// Block: 256 threads = 8 waves. Tile M=128, N=64, K=32. Double-buffered LDS.
// ---------------------------------------------------------------------------
#define BLK_M 128
#define BLK_N 64
#define BLK_K 32
#define KPAD  8     // pitch = 40 halves = 80 B (16B-aligned rows for b128/b64)

template <int INMODE, int OUTMODE>
__global__ __launch_bounds__(256) void gemm_bf16_wmma(
    const void* __restrict__ Xv, const float* __restrict__ W,
    const float* __restrict__ bias, void* __restrict__ Yv,
    int C, int N, int O) {

    __shared__ unsigned short As[2][BLK_M][BLK_K + KPAD];   // W tile (M x K)
    __shared__ unsigned short Bs[2][BLK_N][BLK_K + KPAD];   // X tile (N x K)

    const int tid   = threadIdx.x;
    const int lane  = tid & 31;
    const int wave  = tid >> 5;
    const int lhalf = lane & 15;
    const int hi    = lane >> 4;
    const int b     = blockIdx.z;
    const int mBase = blockIdx.y * BLK_M;
    const int nBase = blockIdx.x * BLK_N;

    const float*          Xf = (const float*)Xv          + (size_t)b * C * N;
    const unsigned short* Xh = (const unsigned short*)Xv + (size_t)b * N * C;
    const bool fullTile = (mBase + BLK_M <= O) && (nBase + BLK_N <= N);
    const int  numK = C / BLK_K;        // all C here are multiples of 32
    const bool useAsync = (INMODE == 1) && fullTile;

    // A tile: always f32 weights -> bf16 (vectorized when fullTile)
    auto stageA_fast = [&](int kc, int buf) {
#pragma unroll
        for (int t = 0; t < 4; ++t) {               // 1024 float4 / 256 thr
            int idx = t * 256 + tid;
            int row = idx >> 3, kq = idx & 7;
            float4 w4 = *(const float4*)(W + (size_t)(mBase + row) * C + kc + kq * 4);
            *(uint2*)&As[buf][row][kq * 4] = pack4_bf16(w4);
        }
    };
    auto stageA_edge = [&](int kc, int buf) {
#pragma unroll
        for (int t = 0; t < 16; ++t) {
            int idx = t * 256 + tid;
            int i = idx >> 5, kk = idx & 31;
            int o  = mBase + i;
            int oc = (o < O) ? o : (O - 1);
            float v = W[(size_t)oc * C + kc + kk];
            As[buf][i][kk] = f32_to_bf16_rne((o < O) ? v : 0.f);
        }
    };
    // B tile staging
    auto stageB = [&](int kc, int buf) {
        if (INMODE == 1) {
            if (fullTile) {
                // Pure byte copy: CDNA5 async global->LDS, 16B per lane,
                // wave w moves rows w*8 .. w*8+7 (64B of bf16 K-data each).
                int nn = wave * 8 + (lane >> 2);
                int ch = lane & 3;
                const unsigned short* gsrc =
                    Xh + (size_t)(nBase + nn) * C + kc;               // 64B-aligned
                unsigned long long ga =
                    (unsigned long long)(size_t)gsrc + (unsigned)(ch * 16);
                unsigned int lo =
                    (unsigned int)(size_t)&Bs[buf][nn][0] + (unsigned)(ch * 16);
                asm volatile("global_load_async_to_lds_b128 %0, %1, off"
                             :: "v"(lo), "v"(ga) : "memory");
            } else {
#pragma unroll
                for (int t = 0; t < 8; ++t) {
                    int idx = t * 256 + tid;
                    int nn = idx >> 5, kk = idx & 31;
                    int n   = nBase + nn;
                    int ncl = (n < N) ? n : (N - 1);
                    unsigned short raw = Xh[(size_t)ncl * C + kc + kk];
                    Bs[buf][nn][kk] = (n < N) ? raw : (unsigned short)0;
                }
            }
        } else {
            if (fullTile) {
#pragma unroll
                for (int t = 0; t < 2; ++t) {       // 4 coalesced b32 / thr / iter
                    int idx = t * 256 + tid;
                    int nn = idx & 63, kq = idx >> 6;    // kq: 0..3 then 4..7
                    const float* src = Xf + (size_t)(kc + kq * 4) * N + nBase + nn;
                    float4 x4;
                    x4.x = src[0];
                    x4.y = src[(size_t)N];
                    x4.z = src[(size_t)2 * N];
                    x4.w = src[(size_t)3 * N];
                    *(uint2*)&Bs[buf][nn][kq * 4] = pack4_bf16(x4);
                }
                if (kc + 2 * BLK_K < C)             // L2 prefetch of streamed operand
                    __builtin_prefetch(Xf + (size_t)(kc + 2 * BLK_K) * N + nBase + (tid & 63), 0, 0);
            } else {
#pragma unroll
                for (int t = 0; t < 8; ++t) {
                    int idx = t * 256 + tid;
                    int kk = idx >> 6, nn = idx & 63;
                    int n   = nBase + nn;
                    int ncl = (n < N) ? n : (N - 1);
                    float v = Xf[(size_t)(kc + kk) * N + ncl];
                    Bs[buf][nn][kk] = f32_to_bf16_rne((n < N) ? v : 0.f);
                }
            }
        }
    };

    v8f acc[4] = {};

    if (fullTile) stageA_fast(0, 0); else stageA_edge(0, 0);
    stageB(0, 0);
    if (useAsync) asm volatile("s_wait_asynccnt 0x0" ::: "memory");
    __syncthreads();

    for (int kb = 0; kb < numK; ++kb) {
        const int cur = kb & 1;
        const bool more = (kb + 1 < numK);
        if (more) {                          // overlap next-chunk staging with WMMA
            const int kc2 = (kb + 1) * BLK_K;
            if (fullTile) stageA_fast(kc2, cur ^ 1); else stageA_edge(kc2, cur ^ 1);
            stageB(kc2, cur ^ 1);
        }

        // A fragment (16-bit A 16x32): lanes 0-15 K in {0..15}; 16-31 {16..31}
        v16bf afrag;
        const unsigned short* Ar = &As[cur][wave * 16 + lhalf][0];
#pragma unroll
        for (int h = 0; h < 8; ++h) {
            afrag[h]     = __builtin_bit_cast(__bf16, Ar[hi * 16 + h]);
            afrag[h + 8] = __builtin_bit_cast(__bf16, Ar[hi * 16 + 8 + h]);
        }
        // B fragments (16-bit B 32x16): lane n = lhalf, K = hi*16 + h
#pragma unroll
        for (int s = 0; s < 4; ++s) {
            const unsigned short* Br = &Bs[cur][s * 16 + lhalf][hi * 16];
            v16bf bfrag;
#pragma unroll
            for (int h = 0; h < 16; ++h)
                bfrag[h] = __builtin_bit_cast(__bf16, Br[h]);
            acc[s] = __builtin_amdgcn_wmma_f32_16x16x32_bf16(
                false, afrag, false, bfrag, (short)0, acc[s], false, false);
        }
        if (more && useAsync) asm volatile("s_wait_asynccnt 0x0" ::: "memory");
        __syncthreads();
    }

    // ---- epilogue: D layout -> m = wave*16 + hi*8 + r, n = s*16 + lhalf ----
    const int m0 = mBase + wave * 16 + hi * 8;      // 8 consecutive m per lane
#pragma unroll
    for (int s = 0; s < 4; ++s) {
        int n = nBase + s * 16 + lhalf;
        if (n >= N) continue;
        if (OUTMODE == 0) {                         // bf16 NHWC, one b128 store
            unsigned px[4];
#pragma unroll
            for (int r = 0; r < 8; r += 2) {
                unsigned short l16 = f32_to_bf16_rne(acc[s][r]     + bias[m0 + r]);
                unsigned short h16 = f32_to_bf16_rne(acc[s][r + 1] + bias[m0 + r + 1]);
                px[r >> 1] = (unsigned)l16 | ((unsigned)h16 << 16);
            }
            unsigned short* dst = (unsigned short*)Yv + ((size_t)b * N + n) * O + m0;
            *(uint4*)dst = make_uint4(px[0], px[1], px[2], px[3]);
        } else {                                    // f32 head permute
#pragma unroll
            for (int r = 0; r < 8; ++r) {
                int m = m0 + r;
                if (m >= O) continue;
                float val = acc[s][r] + bias[m];
                int a85 = m / 85, ch = m - a85 * 85;
                ((float*)Yv)[(((size_t)b * 3 + a85) * N + n) * 85 + ch] = val;
            }
        }
    }
}

// ---------------------------------------------------------------------------
// CLLA attention, all-bf16 I/O. One wave32 per coarse pixel; lane owns 4
// channels. q,f2,f: [b,6400,128] NHWC; k,v: [b,25600,128] NHWC fine grid.
// ---------------------------------------------------------------------------
__global__ __launch_bounds__(256) void clla_attention(
    const unsigned short* __restrict__ q, const unsigned short* __restrict__ k,
    const unsigned short* __restrict__ v, const unsigned short* __restrict__ f2,
    unsigned short* __restrict__ f) {

    const int lane = threadIdx.x & 31;
    const int wave = threadIdx.x >> 5;
    const int p    = blockIdx.x * 8 + wave;      // 0 .. 51199
    const int b    = p / 6400;
    const int n    = p - b * 6400;
    const int h    = n / 80;
    const int w    = n - h * 80;

    uint2 q2 = ((const uint2*)(q + (size_t)p * 128))[lane];   // 4 bf16 channels
    float qx, qy, qz, qw;
    unpack2_bf16(q2.x, qx, qy);
    unpack2_bf16(q2.y, qz, qw);

    float  dots[4];
    size_t fbase[4];
#pragma unroll
    for (int t = 0; t < 4; ++t) {                // t = 2*i + j
        int y = 2 * h + (t >> 1);
        int x = 2 * w + (t & 1);
        size_t fp = ((size_t)b * 25600 + (size_t)y * 160 + x) * 128;
        fbase[t] = fp;
        uint2 k2 = *(const uint2*)(k + fp + lane * 4);
        float kx, ky, kz, kw;
        unpack2_bf16(k2.x, kx, ky);
        unpack2_bf16(k2.y, kz, kw);
        float part = qx * kx + qy * ky + qz * kz + qw * kw;
#pragma unroll
        for (int off = 16; off > 0; off >>= 1)
            part += __shfl_xor(part, off, 32);
        dots[t] = part * 0.5f;                   // / range
    }
    float mean = 0.25f * (dots[0] + dots[1] + dots[2] + dots[3]);
    float irr[4], mx = -3.0e38f;
#pragma unroll
    for (int t = 0; t < 4; ++t) {
        irr[t] = 2.f * mean - dots[t];
        mx = fmaxf(mx, irr[t]);
    }
    float e[4], sum = 0.f;
#pragma unroll
    for (int t = 0; t < 4; ++t) { e[t] = __expf(irr[t] - mx); sum += e[t]; }
    float inv = 1.f / sum;

    float ox = 0.f, oy = 0.f, oz = 0.f, ow = 0.f;
#pragma unroll
    for (int t = 0; t < 4; ++t) {
        float a = e[t] * inv;
        uint2 v2 = *(const uint2*)(v + fbase[t] + lane * 4);
        float vx, vy, vz, vw;
        unpack2_bf16(v2.x, vx, vy);
        unpack2_bf16(v2.y, vz, vw);
        ox += a * vx; oy += a * vy; oz += a * vz; ow += a * vw;
    }
    uint2 f22 = ((const uint2*)(f2 + (size_t)p * 128))[lane];
    float fx, fy, fz, fw;
    unpack2_bf16(f22.x, fx, fy);
    unpack2_bf16(f22.y, fz, fw);
    float4 outv;
    outv.x = (ox + fx) * 0.5f;
    outv.y = (oy + fy) * 0.5f;
    outv.z = (oz + fz) * 0.5f;
    outv.w = (ow + fw) * 0.5f;
    ((uint2*)(f + (size_t)p * 128))[lane] = pack4_bf16(outv);
}

// ---------------------------------------------------------------------------
// Host orchestration (all on `stream`, graph-capture safe).
// ---------------------------------------------------------------------------
extern "C" void kernel_launch(void* const* d_in, const int* in_sizes, int n_in,
                              void* d_out, int out_size, void* d_ws, size_t ws_size,
                              hipStream_t stream) {
    const float* x0      = (const float*)d_in[0];   // [8,128,160,160]
    const float* x1      = (const float*)d_in[1];   // [8,256,80,80]
    const float* x2      = (const float*)d_in[2];   // [8,512,40,40]
    const float* x3      = (const float*)d_in[3];   // [8,1024,20,20]
    const float* w_conv1 = (const float*)d_in[4];
    const float* b_conv1 = (const float*)d_in[5];
    const float* w_conv2 = (const float*)d_in[6];
    const float* b_conv2 = (const float*)d_in[7];
    const float* wq      = (const float*)d_in[8];
    const float* bq      = (const float*)d_in[9];
    const float* wk      = (const float*)d_in[10];
    const float* bk      = (const float*)d_in[11];
    const float* wv      = (const float*)d_in[12];
    const float* bv      = (const float*)d_in[13];
    const float* w_det   = (const float*)d_in[14];
    const float* b_det   = (const float*)d_in[15];
    const float* wm0     = (const float*)d_in[16];
    const float* bm0     = (const float*)d_in[17];
    const float* wm1     = (const float*)d_in[18];
    const float* bm1     = (const float*)d_in[19];

    float* out = (float*)d_out;

    // workspace: f32 folded weights, then bf16 NHWC intermediates (16B aligned)
    float* ws      = (float*)d_ws;
    float* wk_eff  = ws;                        // 128*128
    float* wv_eff  = wk_eff + 16384;
    float* bk_eff  = wv_eff + 16384;            // 128
    float* bv_eff  = bk_eff + 128;
    unsigned short* hws  = (unsigned short*)(bv_eff + 128);  // 132096 B offset
    unsigned short* f2   = hws;                              // [8,6400,128]
    unsigned short* qbuf = f2   + (size_t)8 * 6400 * 128;
    unsigned short* kbuf = qbuf + (size_t)8 * 6400 * 128;    // [8,25600,128]
    unsigned short* vbuf = kbuf + (size_t)8 * 25600 * 128;
    unsigned short* fbuf = vbuf + (size_t)8 * 25600 * 128;   // [8,6400,128]

    // 1) fold w_conv1 into wk / wv so f1 is never materialized
    fuse_weights<<<128, 128, 0, stream>>>(wk, w_conv1, bk, b_conv1, wk_eff, bk_eff);
    fuse_weights<<<128, 128, 0, stream>>>(wv, w_conv1, bv, b_conv1, wv_eff, bv_eff);

    // 2) f2 = conv1x1(x1, w_conv2) -> bf16 NHWC   (C=256, N=6400, O=128)
    gemm_bf16_wmma<0, 0><<<dim3(100, 1, 8), 256, 0, stream>>>(
        x1, w_conv2, b_conv2, f2, 256, 6400, 128);

    // 3) k,v = conv1x1(x0, folded) -> bf16 NHWC   (C=128, N=25600, O=128)
    gemm_bf16_wmma<0, 0><<<dim3(400, 1, 8), 256, 0, stream>>>(
        x0, wk_eff, bk_eff, kbuf, 128, 25600, 128);
    gemm_bf16_wmma<0, 0><<<dim3(400, 1, 8), 256, 0, stream>>>(
        x0, wv_eff, bv_eff, vbuf, 128, 25600, 128);

    // 4) q = linear(f2, wq) -> bf16 NHWC          (C=128, N=6400, O=128)
    //    bf16 input => async global->LDS staging
    gemm_bf16_wmma<1, 0><<<dim3(100, 1, 8), 256, 0, stream>>>(
        f2, wq, bq, qbuf, 128, 6400, 128);

    // 5) attention + residual -> f (bf16 NHWC)
    clla_attention<<<6400, 256, 0, stream>>>(qbuf, kbuf, vbuf, f2, fbuf);

    // 6) detection heads (f32), permuted straight into d_out
    gemm_bf16_wmma<1, 1><<<dim3(100, 2, 8), 256, 0, stream>>>(
        fbuf, w_det, b_det, out, 128, 6400, 255);                 // p0
    gemm_bf16_wmma<0, 1><<<dim3(25, 2, 8), 256, 0, stream>>>(
        x2, wm0, bm0, out + 13056000, 512, 1600, 255);            // p1
    gemm_bf16_wmma<0, 1><<<dim3(7, 2, 8), 256, 0, stream>>>(
        x3, wm1, bm1, out + 16320000, 1024, 400, 255);            // p2
}
